// MultiHeadSelfAttention_1468878815342
// MI455X (gfx1250) — matrix operand order repaired and checked by reading, not compiled, not run on hardware
//
#include <hip/hip_runtime.h>

// ---------------------------------------------------------------------------
// Multi-head self-attention for MI455X (gfx1250, wave32, WMMA).
// All matmuls run on v_wmma_f32_16x16x32_bf16 (f32 accumulate).
// Round 2: software-pipelined fragment loads, 32x64 wave tile in the GEMM
// (8 WMMA per 6 fragment loads), V-fragment loads hoisted over the softmax VALU work.
// ---------------------------------------------------------------------------

#define D_MODEL 1024
#define SEQ     2048
#define HEADS   16
#define DK      64
#define BATCH   2
#define MROWS   (BATCH * SEQ)   // 4096 rows for the projection GEMMs

typedef __bf16        v16bf __attribute__((ext_vector_type(16)));
typedef float         v8f   __attribute__((ext_vector_type(8)));
typedef unsigned int  u32x4 __attribute__((ext_vector_type(4)));

union Frag16 { u32x4 u[2]; v16bf v; };

// float -> bf16 bits with round-to-nearest-even
__device__ __forceinline__ unsigned short f2bf(float f) {
    unsigned int u = __builtin_bit_cast(unsigned int, f);
    u += 0x7FFFu + ((u >> 16) & 1u);
    return (unsigned short)(u >> 16);
}

// ---------------------------------------------------------------------------
// WMMA fragment loaders (layouts per CDNA5 ISA §7.12.2, wave32).
//
// A-matrix 16x32 bf16 (MxK): lane l -> M = l%16; lanes 0-15 hold K pairs
// {0..7,16..23}, lanes 16-31 hold {8..15,24..31}.  Row-major memory => two
// contiguous 16-byte loads per lane.
// ---------------------------------------------------------------------------
__device__ __forceinline__ v16bf load_fragA(const unsigned short* base, int stride,
                                            int rowBase, int k0) {
    int lane = threadIdx.x & 31;
    int r    = rowBase + (lane & 15);
    int kk   = k0 + ((lane & 16) ? 8 : 0);
    const unsigned short* p = base + (size_t)r * stride + kk;
    Frag16 f;
    f.u[0] = *reinterpret_cast<const u32x4*>(p);
    f.u[1] = *reinterpret_cast<const u32x4*>(p + 16);
    return f.v;
}

// B-matrix 32x16 bf16 (KxN): lane l -> N = l%16; lanes 0-15 hold K {0..7,16..23},
// lanes 16-31 hold {8..15,24..31} (same K order as A).  Row-major [K,N] memory =>
// 16 strided bf16 gathers per lane.
__device__ __forceinline__ v16bf load_fragB(const unsigned short* base, int stride,
                                            int kBase, int nBase) {
    int lane = threadIdx.x & 31;
    int k    = kBase + ((lane & 16) ? 8 : 0);
    const unsigned short* p = base + (size_t)k * stride + nBase + (lane & 15);
    Frag16 f;
#pragma unroll
    for (int s = 0; s < 8; ++s) {
        f.v[s]     = __builtin_bit_cast(__bf16, p[(size_t)s * stride]);
        f.v[s + 8] = __builtin_bit_cast(__bf16, p[(size_t)(s + 16) * stride]);
    }
    return f.v;
}

__device__ __forceinline__ v8f wmma_bf16(v16bf a, v16bf b, v8f c) {
    // (neg_a, A, neg_b, B, c_mod, C, reuse_a, reuse_b)
    return __builtin_amdgcn_wmma_f32_16x16x32_bf16(false, a, false, b,
                                                   (short)0, c, false, false);
}

// ---------------------------------------------------------------------------
// fp32 -> bf16 elementwise convert
// ---------------------------------------------------------------------------
__global__ void convert_f32_bf16(const float* __restrict__ in,
                                 unsigned short* __restrict__ out, int n) {
    int i = blockIdx.x * blockDim.x + threadIdx.x;
    if (i < n) out[i] = f2bf(in[i]);
}

// ---------------------------------------------------------------------------
// GEMM: out[m,n] = sum_k A[m,k]*W[k,n] + bias[n]
//   A:  bf16 [M, 1024] row-major, W: bf16 [1024, 1024] row-major.
//   Block = 256 threads (8 waves); block tile 256x64; wave tile 32x64:
//   per k-slab: 2 A frags + 4 B frags -> 8 WMMA (B reused across both rows).
// mode 0: bf16 out at [B,H,S,dk]   (Q, V)
// mode 1: bf16 out at [B,H,dk,S]   (K transposed)
// mode 2: f32  out at [M,N]        (final projection)
// ---------------------------------------------------------------------------
__global__ void gemm_bf16_wmma(const unsigned short* __restrict__ A,
                               const unsigned short* __restrict__ W,
                               const float* __restrict__ bias,
                               void* __restrict__ out, int mode) {
    const int K = D_MODEL, N = D_MODEL;
    int wave = threadIdx.x >> 5;
    int lane = threadIdx.x & 31;
    int rowBase = blockIdx.x * 256 + wave * 32;   // wave owns rows rowBase..+31
    int colBase = blockIdx.y * 64;

    v8f acc0[4], acc1[4];
#pragma unroll
    for (int t = 0; t < 4; ++t)
#pragma unroll
        for (int i = 0; i < 8; ++i) { acc0[t][i] = 0.0f; acc1[t][i] = 0.0f; }

    for (int k0 = 0; k0 < K; k0 += 32) {
        if (k0 + 32 < K) {
            // prefetch next A slab (global_prefetch_b8)
            __builtin_prefetch(A + (size_t)(rowBase + (lane & 15)) * K + k0 + 32, 0, 3);
        }
        // Issue ALL fragment loads for this k-slab first so the scheduler can
        // drain loadcnt progressively instead of s_wait_loadcnt 0 per WMMA.
        v16bf a0 = load_fragA(A, K, rowBase,      k0);
        v16bf a1 = load_fragA(A, K, rowBase + 16, k0);
        v16bf b[4];
#pragma unroll
        for (int t = 0; t < 4; ++t) b[t] = load_fragB(W, N, k0, colBase + t * 16);
#pragma unroll
        for (int t = 0; t < 4; ++t) acc0[t] = wmma_bf16(a0, b[t], acc0[t]);
#pragma unroll
        for (int t = 0; t < 4; ++t) acc1[t] = wmma_bf16(a1, b[t], acc1[t]);
    }

    int hi = (lane & 16) ? 8 : 0;
#pragma unroll
    for (int half = 0; half < 2; ++half) {
        const v8f* acc = half ? acc1 : acc0;
        int rb = rowBase + half * 16;
#pragma unroll
        for (int t = 0; t < 4; ++t) {
            int n = colBase + t * 16 + (lane & 15);
            float bn = bias[n];
#pragma unroll
            for (int v = 0; v < 8; ++v) {
                int   m   = rb + hi + v;
                float val = acc[t][v] + bn;
                if (mode == 2) {
                    reinterpret_cast<float*>(out)[(size_t)m * N + n] = val;
                } else {
                    int b2 = m >> 11, s = m & (SEQ - 1);
                    int h  = n >> 6,  d = n & (DK - 1);
                    size_t idx = (mode == 0)
                        ? ((size_t)(b2 * HEADS + h) * SEQ + s) * DK + d
                        : ((size_t)(b2 * HEADS + h) * DK  + d) * SEQ + s;
                    reinterpret_cast<unsigned short*>(out)[idx] = f2bf(val);
                }
            }
        }
    }
}

// ---------------------------------------------------------------------------
// Flash attention per (b,h): Q [S,dk] bf16, Kt [dk,S] bf16, V [S,dk] bf16.
// Block = 256 threads = 8 waves; each wave owns 16 query rows, streams keys in
// 64-wide chunks with online softmax.  P is restaged via a 2KB per-wave LDS
// tile to convert the C-layout scores into A-fragments for P x V.
// V fragments are loaded BEFORE the softmax VALU work so the exp/shuffle
// reductions hide the memory latency of the P*V operands.
// ---------------------------------------------------------------------------
__global__ void flash_attention_wmma(const unsigned short* __restrict__ Qh,
                                     const unsigned short* __restrict__ Kt,
                                     const unsigned short* __restrict__ Vh,
                                     unsigned short* __restrict__ Cc) {
    __shared__ unsigned short Pbuf[8][16 * 64];   // 16 KB

    int wave = threadIdx.x >> 5;
    int lane = threadIdx.x & 31;
    int hi   = (lane & 16) ? 8 : 0;

    int bh    = blockIdx.x;                  // b*HEADS + h
    int qBase = blockIdx.y * 128 + wave * 16;

    const unsigned short* Qp = Qh + (size_t)bh * SEQ * DK;  // [S, dk]
    const unsigned short* Kp = Kt + (size_t)bh * DK * SEQ;  // [dk, S]
    const unsigned short* Vp = Vh + (size_t)bh * SEQ * DK;  // [S, dk]
    unsigned short* P = &Pbuf[wave][0];

    // Q A-fragments for the whole key loop (dk split 0..31 / 32..63)
    v16bf qa0 = load_fragA(Qp, DK, qBase, 0);
    v16bf qa1 = load_fragA(Qp, DK, qBase, 32);

    const float scale = 0.125f;              // 1/sqrt(dk), dk=64
    float m_i[8], l_i[8];
    v8f acc[4];
#pragma unroll
    for (int v = 0; v < 8; ++v) { m_i[v] = -1e30f; l_i[v] = 0.0f; }
#pragma unroll
    for (int t = 0; t < 4; ++t)
#pragma unroll
        for (int i = 0; i < 8; ++i) acc[t][i] = 0.0f;

    for (int key0 = 0; key0 < SEQ; key0 += 64) {
        // ---- scores: load all 8 K fragments, then 8 WMMAs ----------------
        v16bf kb0[4], kb1[4];
#pragma unroll
        for (int t = 0; t < 4; ++t) {
            kb0[t] = load_fragB(Kp, SEQ, 0,  key0 + t * 16);
            kb1[t] = load_fragB(Kp, SEQ, 32, key0 + t * 16);
        }
        float sc[4][8];
#pragma unroll
        for (int t = 0; t < 4; ++t) {
            v8f c = {};
            c = wmma_bf16(qa0, kb0[t], c);
            c = wmma_bf16(qa1, kb1[t], c);
#pragma unroll
            for (int v = 0; v < 8; ++v) sc[t][v] = c[v] * scale;
        }

        // ---- issue V fragment loads now; softmax VALU hides the latency --
        v16bf vb0[4], vb1[4];
#pragma unroll
        for (int t = 0; t < 4; ++t) {
            vb0[t] = load_fragB(Vp, DK, key0,      t * 16);
            vb1[t] = load_fragB(Vp, DK, key0 + 32, t * 16);
        }

        // ---- online softmax: row max over 64 keys ------------------------
        float mnew[8];
#pragma unroll
        for (int v = 0; v < 8; ++v) {
            float mx = sc[0][v];
#pragma unroll
            for (int t = 1; t < 4; ++t) mx = fmaxf(mx, sc[t][v]);
            // butterfly across the 16-lane half (rows hi..hi+7 live here)
            mx = fmaxf(mx, __shfl_xor(mx, 1));
            mx = fmaxf(mx, __shfl_xor(mx, 2));
            mx = fmaxf(mx, __shfl_xor(mx, 4));
            mx = fmaxf(mx, __shfl_xor(mx, 8));
            mnew[v] = fmaxf(m_i[v], mx);
        }

        float rs[8];
#pragma unroll
        for (int v = 0; v < 8; ++v) rs[v] = 0.0f;
#pragma unroll
        for (int t = 0; t < 4; ++t) {
            int n = t * 16 + (lane & 15);
#pragma unroll
            for (int v = 0; v < 8; ++v) {
                float p = __expf(sc[t][v] - mnew[v]);
                rs[v] += p;
                P[(hi + v) * 64 + n] = f2bf(p);   // stage P tile in LDS
            }
        }
#pragma unroll
        for (int v = 0; v < 8; ++v) {
            rs[v] += __shfl_xor(rs[v], 1);
            rs[v] += __shfl_xor(rs[v], 2);
            rs[v] += __shfl_xor(rs[v], 4);
            rs[v] += __shfl_xor(rs[v], 8);
            float corr = __expf(m_i[v] - mnew[v]);
            l_i[v] = l_i[v] * corr + rs[v];
            m_i[v] = mnew[v];
#pragma unroll
            for (int t = 0; t < 4; ++t) acc[t][v] *= corr;
        }

        __syncthreads();  // uniform; DS is in-order per wave, barrier for safety

        // ---- O += P * V --------------------------------------------------
        v16bf pa0 = load_fragA(P, 64, 0, 0);    // keys key0+0..31
        v16bf pa1 = load_fragA(P, 64, 0, 32);   // keys key0+32..63
#pragma unroll
        for (int t = 0; t < 4; ++t) acc[t] = wmma_bf16(pa0, vb0[t], acc[t]);
#pragma unroll
        for (int t = 0; t < 4; ++t) acc[t] = wmma_bf16(pa1, vb1[t], acc[t]);
        __syncthreads();
    }

    // ---- normalize and write concat [B, S, H*dk] as bf16 ----------------
    int b = bh >> 4, h = bh & (HEADS - 1);
#pragma unroll
    for (int t = 0; t < 4; ++t) {
        int d = t * 16 + (lane & 15);
#pragma unroll
        for (int v = 0; v < 8; ++v) {
            int s = qBase + hi + v;
            float val = acc[t][v] / l_i[v];
            Cc[(((size_t)b * SEQ + s) * HEADS + h) * DK + d] = f2bf(val);
        }
    }
}

// ---------------------------------------------------------------------------
// Launch
// ---------------------------------------------------------------------------
extern "C" void kernel_launch(void* const* d_in, const int* in_sizes, int n_in,
                              void* d_out, int out_size, void* d_ws, size_t ws_size,
                              hipStream_t stream) {
    const float* X  = (const float*)d_in[0];
    const float* Wq = (const float*)d_in[1];
    const float* bq = (const float*)d_in[2];
    const float* Wk = (const float*)d_in[3];
    const float* bk = (const float*)d_in[4];
    const float* Wv = (const float*)d_in[5];
    const float* bv = (const float*)d_in[6];
    const float* Wo = (const float*)d_in[7];
    const float* bo = (const float*)d_in[8];

    char* ws = (char*)d_ws;
    const size_t MB = 1024u * 1024u;
    unsigned short* Xb  = (unsigned short*)(ws +  0 * MB);  // 8 MB  [4096,1024]
    unsigned short* Wqb = (unsigned short*)(ws +  8 * MB);  // 2 MB  [1024,1024]
    unsigned short* Wkb = (unsigned short*)(ws + 10 * MB);
    unsigned short* Wvb = (unsigned short*)(ws + 12 * MB);
    unsigned short* Wob = (unsigned short*)(ws + 14 * MB);
    unsigned short* Qh  = (unsigned short*)(ws + 16 * MB);  // 8 MB  [B,H,S,dk]
    unsigned short* Kt  = (unsigned short*)(ws + 24 * MB);  // 8 MB  [B,H,dk,S]
    unsigned short* Vh  = (unsigned short*)(ws + 32 * MB);  // 8 MB  [B,H,S,dk]
    unsigned short* Cc  = (unsigned short*)(ws + 40 * MB);  // 8 MB  [B,S,D]
    // total workspace use: 48 MiB

    const int nX = MROWS * D_MODEL;
    const int nW = D_MODEL * D_MODEL;
    convert_f32_bf16<<<(nX + 255) / 256, 256, 0, stream>>>(X,  Xb,  nX);
    convert_f32_bf16<<<(nW + 255) / 256, 256, 0, stream>>>(Wq, Wqb, nW);
    convert_f32_bf16<<<(nW + 255) / 256, 256, 0, stream>>>(Wk, Wkb, nW);
    convert_f32_bf16<<<(nW + 255) / 256, 256, 0, stream>>>(Wv, Wvb, nW);
    convert_f32_bf16<<<(nW + 255) / 256, 256, 0, stream>>>(Wo, Wob, nW);

    dim3 ggrid(MROWS / 256, D_MODEL / 64);   // 16 x 16
    gemm_bf16_wmma<<<ggrid, 256, 0, stream>>>(Xb, Wqb, bq, (void*)Qh, 0);
    gemm_bf16_wmma<<<ggrid, 256, 0, stream>>>(Xb, Wkb, bk, (void*)Kt, 1);
    gemm_bf16_wmma<<<ggrid, 256, 0, stream>>>(Xb, Wvb, bv, (void*)Vh, 0);

    flash_attention_wmma<<<dim3(BATCH * HEADS, SEQ / 128), 256, 0, stream>>>(Qh, Kt, Vh, Cc);

    gemm_bf16_wmma<<<ggrid, 256, 0, stream>>>(Cc, Wob, bo, d_out, 2);
}
